// simgnn_71768903516462
// MI455X (gfx1250) — matrix operand order repaired and checked by reading, not compiled, not run on hardware
//
#include <hip/hip_runtime.h>
#include <hip/hip_bf16.h>

// ---------------------------------------------------------------------------
// Types for CDNA5 WMMA (wave32): D(16x16 f32) = A(16x32 bf16) x B(32x16 bf16) + C
// ---------------------------------------------------------------------------
typedef __attribute__((ext_vector_type(16))) __bf16 v16bf;
typedef __attribute__((ext_vector_type(8)))  float  v8f;

// A fragment: 16x32 bf16, row-major source A[row][k] (f32), lda floats.
// Layout: lanes 0-15 hold M=lane, K = {0..15};
// lanes 16-31 hold K = {16..31} (matches B fragment).
__device__ __forceinline__ v16bf load_a_frag(const float* __restrict__ A, int row0,
                                             int lda, int kbase, int lane) {
    int m  = lane & 15;
    int kh = (lane >> 4) * 16;
    const float* p = A + (size_t)(row0 + m) * lda + kbase;
    v16bf a;
#pragma unroll
    for (int i = 0; i < 8; ++i) a[i]     = static_cast<__bf16>(p[kh + i]);
#pragma unroll
    for (int i = 0; i < 8; ++i) a[8 + i] = static_cast<__bf16>(p[kh + 8 + i]);
    return a;
}

// B fragment: 32x16 bf16 from COLUMN-MAJOR packed bf16 weights Wt[col][k]
// (K contiguous per column). Lane holds column (col0 + lane&15);
// lanes 0-15 take K = kbase..kbase+15, lanes 16-31 K = kbase+16..kbase+31.
// Per-lane slice is 16 contiguous bf16 = 32B -> two global_load_b128.
__device__ __forceinline__ v16bf load_b_frag(const __bf16* __restrict__ Wt, int K,
                                             int kbase, int col0, int lane) {
    int col = col0 + (lane & 15);
    const __bf16* p = Wt + (size_t)col * K + kbase + ((lane >> 4) << 4);
    return *(const v16bf*)p;
}

// C/D: reg v holds D[v + 8*(lane>>4)][lane&15]
__device__ __forceinline__ void store_c_frag(float* __restrict__ C, int row0, int ldc,
                                             int col0, int lane, v8f c,
                                             const float* __restrict__ bias, bool relu) {
    int col   = col0 + (lane & 15);
    int rbase = row0 + (lane >> 4) * 8;
    float bv = bias ? bias[col] : 0.0f;
#pragma unroll
    for (int v = 0; v < 8; ++v) {
        float val = c[v] + bv;
        if (relu) val = fmaxf(val, 0.0f);
        C[(size_t)(rbase + v) * ldc + col] = val;
    }
}

// ---------------------------------------------------------------------------
// One-shot weight conversion f32 -> bf16 with row->column-major transpose.
//   [0,2048)    Wt_gcn[c][k]  c<32, k<64   from W_gcn[k*32+c]
//   [2048,4096) Wt_ll[l][c][k] l<2,c<32,k<32 from sage_lin_l[l*1024+k*32+c]
//   [4096,6144) Wt_lr likewise
// ---------------------------------------------------------------------------
__global__ void cvt_weights_k(const float* __restrict__ Wg, const float* __restrict__ ll,
                              const float* __restrict__ lr, __bf16* __restrict__ out) {
    int i = blockIdx.x * blockDim.x + threadIdx.x;
    if (i < 2048) {
        int c = i >> 6, k = i & 63;
        out[i] = static_cast<__bf16>(Wg[k * 32 + c]);
    } else if (i < 4096) {
        int j = i - 2048;
        int l = j >> 10, r = j & 1023;
        int c = r >> 5, k = r & 31;
        out[i] = static_cast<__bf16>(ll[l * 1024 + k * 32 + c]);
    } else if (i < 6144) {
        int j = i - 4096;
        int l = j >> 10, r = j & 1023;
        int c = r >> 5, k = r & 31;
        out[i] = static_cast<__bf16>(lr[l * 1024 + k * 32 + c]);
    }
}

// ---------------------------------------------------------------------------
// GEMM 1: C[N,32] = A[N,64] @ W[64,32]   (no bias/relu; GCN pre-aggregation)
// one wave per 16-row tile, 2 col tiles x 2 K-chunks = 4 WMMAs, A reused
// ---------------------------------------------------------------------------
__global__ void gemm_k64(const float* __restrict__ A, const __bf16* __restrict__ Wt,
                         float* __restrict__ C, int nrows) {
    int wave = blockIdx.x * (blockDim.x >> 5) + (threadIdx.x >> 5);
    int lane = threadIdx.x & 31;
    int row0 = wave * 16;
    if (row0 >= nrows) return;
    v8f acc0 = {}; v8f acc1 = {};
#pragma unroll
    for (int kc = 0; kc < 2; ++kc) {
        v16bf a  = load_a_frag(A, row0, 64, kc * 32, lane);
        v16bf b0 = load_b_frag(Wt, 64, kc * 32, 0,  lane);
        v16bf b1 = load_b_frag(Wt, 64, kc * 32, 16, lane);
        acc0 = __builtin_amdgcn_wmma_f32_16x16x32_bf16(false, a, false, b0, (short)0, acc0, false, false);
        acc1 = __builtin_amdgcn_wmma_f32_16x16x32_bf16(false, a, false, b1, (short)0, acc1, false, false);
    }
    store_c_frag(C, row0, 32, 0,  lane, acc0, nullptr, false);
    store_c_frag(C, row0, 32, 16, lane, acc1, nullptr, false);
}

// ---------------------------------------------------------------------------
// Fused SAGE GEMM: out[N,32] = relu( mean[N,32]@Wl + h[N,32]@Wr + bias )
// ---------------------------------------------------------------------------
__global__ void sage_gemm(const float* __restrict__ mean, const float* __restrict__ h,
                          const __bf16* __restrict__ Wlt, const __bf16* __restrict__ Wrt,
                          const float* __restrict__ bias, float* __restrict__ out, int nrows) {
    int wave = blockIdx.x * (blockDim.x >> 5) + (threadIdx.x >> 5);
    int lane = threadIdx.x & 31;
    int row0 = wave * 16;
    if (row0 >= nrows) return;
    v8f acc0 = {}; v8f acc1 = {};
    {
        v16bf a  = load_a_frag(mean, row0, 32, 0, lane);
        v16bf b0 = load_b_frag(Wlt, 32, 0, 0,  lane);
        v16bf b1 = load_b_frag(Wlt, 32, 0, 16, lane);
        acc0 = __builtin_amdgcn_wmma_f32_16x16x32_bf16(false, a, false, b0, (short)0, acc0, false, false);
        acc1 = __builtin_amdgcn_wmma_f32_16x16x32_bf16(false, a, false, b1, (short)0, acc1, false, false);
    }
    {
        v16bf a  = load_a_frag(h, row0, 32, 0, lane);
        v16bf b0 = load_b_frag(Wrt, 32, 0, 0,  lane);
        v16bf b1 = load_b_frag(Wrt, 32, 0, 16, lane);
        acc0 = __builtin_amdgcn_wmma_f32_16x16x32_bf16(false, a, false, b0, (short)0, acc0, false, false);
        acc1 = __builtin_amdgcn_wmma_f32_16x16x32_bf16(false, a, false, b1, (short)0, acc1, false, false);
    }
    store_c_frag(out, row0, 32, 0,  lane, acc0, bias, true);
    store_c_frag(out, row0, 32, 16, lane, acc1, bias, true);
}

// ---------------------------------------------------------------------------
// Edge / segment kernels (L2-resident gathers + f32 atomics scatter)
// ---------------------------------------------------------------------------
__global__ void count_dst_k(const int* __restrict__ dst, int E, float* __restrict__ cnt) {
    int i = blockIdx.x * blockDim.x + threadIdx.x;
    if (i < E) atomicAdd(&cnt[dst[i]], 1.0f);
}

__global__ void dinv_k(const float* __restrict__ deg, float* __restrict__ dinv, int N) {
    int n = blockIdx.x * blockDim.x + threadIdx.x;
    if (n < N) dinv[n] = rsqrtf(deg[n] + 1.0f);
}

// GCN messages: agg[dst] += xw[src] * dinv[src]*dinv[dst]; 8 threads/edge x float4
__global__ void gcn_edge_k(const int* __restrict__ src, const int* __restrict__ dst,
                           const float* __restrict__ xw, const float* __restrict__ dinv,
                           float* __restrict__ agg, int E) {
    int idx = blockIdx.x * blockDim.x + threadIdx.x;
    if (idx >= E * 8) return;
    int e = idx >> 3, f = (idx & 7) << 2;
    int s = src[e], d = dst[e];
    float w = dinv[s] * dinv[d];
    float4 v = *(const float4*)(xw + (size_t)s * 32 + f);
    float* p = agg + (size_t)d * 32 + f;
    atomicAdd(p + 0, v.x * w); atomicAdd(p + 1, v.y * w);
    atomicAdd(p + 2, v.z * w); atomicAdd(p + 3, v.w * w);
}

__global__ void sage_edge_k(const int* __restrict__ src, const int* __restrict__ dst,
                            const float* __restrict__ h, float* __restrict__ sum, int E) {
    int idx = blockIdx.x * blockDim.x + threadIdx.x;
    if (idx >= E * 8) return;
    int e = idx >> 3, f = (idx & 7) << 2;
    int s = src[e], d = dst[e];
    float4 v = *(const float4*)(h + (size_t)s * 32 + f);
    float* p = sum + (size_t)d * 32 + f;
    atomicAdd(p + 0, v.x); atomicAdd(p + 1, v.y);
    atomicAdd(p + 2, v.z); atomicAdd(p + 3, v.w);
}

// h = relu(agg + xw*dinv^2 + b)   (in-place over xw)
__global__ void gcn_finish_k(float* __restrict__ h_xw, const float* __restrict__ agg,
                             const float* __restrict__ dinv, const float* __restrict__ bias, int N) {
    int idx = blockIdx.x * blockDim.x + threadIdx.x;
    if (idx >= N * 32) return;
    int n = idx >> 5, c = idx & 31;
    float di = dinv[n];
    h_xw[idx] = fmaxf(agg[idx] + h_xw[idx] * di * di + bias[c], 0.0f);
}

__global__ void sage_mean_k(float* __restrict__ sum, const float* __restrict__ cnt, int N) {
    int idx = blockIdx.x * blockDim.x + threadIdx.x;
    if (idx >= N * 32) return;
    sum[idx] /= fmaxf(cnt[idx >> 5], 1.0f);
}

// ---------------------------------------------------------------------------
// Attention pooling
// ---------------------------------------------------------------------------
__global__ void pool_sum_k(const float* __restrict__ rep, const int* __restrict__ batch,
                           float* __restrict__ sB, float* __restrict__ cntB, int N) {
    int idx = blockIdx.x * blockDim.x + threadIdx.x;
    if (idx >= N * 8) return;
    int n = idx >> 3, f = (idx & 7) << 2;
    int b = batch[n];
    float4 v = *(const float4*)(rep + (size_t)n * 32 + f);
    float* p = sB + (size_t)b * 32 + f;
    atomicAdd(p + 0, v.x); atomicAdd(p + 1, v.y);
    atomicAdd(p + 2, v.z); atomicAdd(p + 3, v.w);
    if (f == 0) atomicAdd(&cntB[b], 1.0f);
}

__global__ void ctx_k(const float* __restrict__ sB, const float* __restrict__ cntB,
                      const float* __restrict__ Wc, float* __restrict__ ctx, int B) {
    int idx = blockIdx.x * blockDim.x + threadIdx.x;
    if (idx >= B * 32) return;
    int b = idx >> 5, c = idx & 31;
    float inv = 1.0f / fmaxf(cntB[b], 1.0f);
    float acc = 0.0f;
#pragma unroll
    for (int k = 0; k < 32; ++k) acc += sB[(size_t)b * 32 + k] * inv * Wc[(size_t)k * 32 + c];
    ctx[idx] = tanhf(acc);
}

__global__ void gate_pool_k(const float* __restrict__ rep, const int* __restrict__ batch,
                            const float* __restrict__ ctx, float* __restrict__ pooled, int N) {
    int n = blockIdx.x * blockDim.x + threadIdx.x;
    if (n >= N) return;
    int b = batch[n];
    const float* r = rep + (size_t)n * 32;
    const float* c = ctx + (size_t)b * 32;
    float dot = 0.0f;
#pragma unroll
    for (int i = 0; i < 32; ++i) dot += r[i] * c[i];
    float gate = 1.0f / (1.0f + __expf(-dot));
    float* p = pooled + (size_t)b * 32;
#pragma unroll
    for (int i = 0; i < 32; ++i) atomicAdd(p + i, gate * r[i]);
}

// ---------------------------------------------------------------------------
// NTN + scorer: one wave per graph, lane = NTN slice k (K==32==wave32)
// ---------------------------------------------------------------------------
__global__ void ntn_score_k(const float* __restrict__ hi, const float* __restrict__ hj,
                            const float* __restrict__ W_ntn, const float* __restrict__ V_ntn,
                            const float* __restrict__ b_ntn, const float* __restrict__ mlp_w,
                            const float* __restrict__ mlp_b, float* __restrict__ out) {
    int b = blockIdx.x;
    int k = threadIdx.x;           // 0..31
    __shared__ float shi[32], shj[32];
    shi[k] = hi[(size_t)b * 32 + k];
    shj[k] = hj[(size_t)b * 32 + k];
    __syncthreads();
    float g = b_ntn[k];
    const float* Wk = W_ntn + (size_t)k * 32 * 32;
#pragma unroll 4
    for (int i = 0; i < 32; ++i) {
        float a = shi[i];
        const float* Wr = Wk + (size_t)i * 32;
#pragma unroll
        for (int j = 0; j < 32; ++j) g += a * Wr[j] * shj[j];
    }
    const float* Vk = V_ntn + (size_t)k * 64;
#pragma unroll
    for (int t = 0; t < 32; ++t) g += Vk[t] * shi[t] + Vk[32 + t] * shj[t];
    float s = g * mlp_w[k];
#pragma unroll
    for (int off = 16; off > 0; off >>= 1) s += __shfl_xor(s, off, 32);
    if (k == 0) out[b] = s + mlp_b[0];
}

// ---------------------------------------------------------------------------
// Host orchestration
// ---------------------------------------------------------------------------
static inline int cdiv(int a, int b) { return (a + b - 1) / b; }

static void run_side(const float* x, const int* ei, const int* batch,
                     const __bf16* Wg_bf, const float* b_gcn,
                     const __bf16* ll_bf, const float* sage_b_l, const __bf16* lr_bf,
                     const float* weight_c,
                     int N, int E, int B,
                     float* bufA, float* bufB, float* bufC, float* deg, float* dinv,
                     float* sB, float* cntB, float* ctx, float* pooled,
                     hipStream_t stream) {
    const int* src = ei;
    const int* dst = ei + E;
    const int tiles = cdiv(N, 16);
    const int gemm_blocks = cdiv(tiles, 4);     // 4 waves/block (128 thr)

    // 1) xw = x @ W_gcn  (WMMA)
    gemm_k64<<<gemm_blocks, 128, 0, stream>>>(x, Wg_bf, bufA, N);

    // 2) degree + dinv
    hipMemsetAsync(deg, 0, (size_t)N * 4, stream);
    count_dst_k<<<cdiv(E, 256), 256, 0, stream>>>(dst, E, deg);
    dinv_k<<<cdiv(N, 256), 256, 0, stream>>>(deg, dinv, N);

    // 3) GCN aggregation + finish (relu)
    hipMemsetAsync(bufB, 0, (size_t)N * 32 * 4, stream);
    gcn_edge_k<<<cdiv(E * 8, 256), 256, 0, stream>>>(src, dst, bufA, dinv, bufB, E);
    gcn_finish_k<<<cdiv(N * 32, 256), 256, 0, stream>>>(bufA, bufB, dinv, b_gcn, N);

    // 4) two SAGE layers
    float* h = bufA;
    float* nxt = bufC;
    for (int l = 0; l < 2; ++l) {
        hipMemsetAsync(bufB, 0, (size_t)N * 32 * 4, stream);
        hipMemsetAsync(deg, 0, (size_t)N * 4, stream);
        sage_edge_k<<<cdiv(E * 8, 256), 256, 0, stream>>>(src, dst, h, bufB, E);
        count_dst_k<<<cdiv(E, 256), 256, 0, stream>>>(dst, E, deg);
        sage_mean_k<<<cdiv(N * 32, 256), 256, 0, stream>>>(bufB, deg, N);
        sage_gemm<<<gemm_blocks, 128, 0, stream>>>(bufB, h,
                                                   ll_bf + (size_t)l * 32 * 32,
                                                   lr_bf + (size_t)l * 32 * 32,
                                                   sage_b_l + (size_t)l * 32, nxt, N);
        float* t = h; h = nxt; nxt = t;
    }

    // 5) attention pooling
    hipMemsetAsync(sB, 0, (size_t)B * 32 * 4, stream);
    hipMemsetAsync(cntB, 0, (size_t)B * 4, stream);
    pool_sum_k<<<cdiv(N * 8, 256), 256, 0, stream>>>(h, batch, sB, cntB, N);
    ctx_k<<<cdiv(B * 32, 256), 256, 0, stream>>>(sB, cntB, weight_c, ctx, B);
    hipMemsetAsync(pooled, 0, (size_t)B * 32 * 4, stream);
    gate_pool_k<<<cdiv(N, 256), 256, 0, stream>>>(h, batch, ctx, pooled, N);
}

extern "C" void kernel_launch(void* const* d_in, const int* in_sizes, int n_in,
                              void* d_out, int out_size, void* d_ws, size_t ws_size,
                              hipStream_t stream) {
    const float* x_i    = (const float*)d_in[0];
    const int*   ei_i   = (const int*)  d_in[1];
    const int*   bat_i  = (const int*)  d_in[2];
    const float* x_j    = (const float*)d_in[3];
    const int*   ei_j   = (const int*)  d_in[4];
    const int*   bat_j  = (const int*)  d_in[5];
    const float* W_gcn  = (const float*)d_in[6];
    const float* b_gcn  = (const float*)d_in[7];
    const float* s_ll   = (const float*)d_in[8];
    const float* s_bl   = (const float*)d_in[9];
    const float* s_lr   = (const float*)d_in[10];
    const float* Wc     = (const float*)d_in[11];
    const float* W_ntn  = (const float*)d_in[12];
    const float* V_ntn  = (const float*)d_in[13];
    const float* b_ntn  = (const float*)d_in[14];
    const float* mlp_w  = (const float*)d_in[15];
    const float* mlp_b  = (const float*)d_in[16];

    const int N = in_sizes[2];          // 100000
    const int E = in_sizes[1] / 2;      // 3200000
    const int B = out_size;             // 512

    float* ws = (float*)d_ws;
    const size_t NH = (size_t)N * 32;
    float* bufA = ws;
    float* bufB = bufA + NH;
    float* bufC = bufB + NH;
    float* deg  = bufC + NH;
    float* dinv = deg + N;
    float* sB   = dinv + N;
    float* cntB = sB + (size_t)B * 32;
    float* ctx  = cntB + B;
    float* pooled_i = ctx + (size_t)B * 32;
    float* pooled_j = pooled_i + (size_t)B * 32;
    __bf16* wbf     = (__bf16*)(pooled_j + (size_t)B * 32);   // 128B-aligned offset
    __bf16* Wg_bf   = wbf;            // 32 cols x 64 k = 2048
    __bf16* ll_bf   = wbf + 2048;     // 2 x (32 cols x 32 k) = 2048
    __bf16* lr_bf   = wbf + 4096;     // 2 x (32 cols x 32 k) = 2048

    cvt_weights_k<<<cdiv(6144, 256), 256, 0, stream>>>(W_gcn, s_ll, s_lr, wbf);

    run_side(x_i, ei_i, bat_i, Wg_bf, b_gcn, ll_bf, s_bl, lr_bf, Wc,
             N, E, B, bufA, bufB, bufC, deg, dinv, sB, cntB, ctx, pooled_i, stream);
    run_side(x_j, ei_j, bat_j, Wg_bf, b_gcn, ll_bf, s_bl, lr_bf, Wc,
             N, E, B, bufA, bufB, bufC, deg, dinv, sB, cntB, ctx, pooled_j, stream);

    ntn_score_k<<<B, 32, 0, stream>>>(pooled_i, pooled_j, W_ntn, V_ntn, b_ntn,
                                      mlp_w, mlp_b, (float*)d_out);
}
